// BitMLP_41652592836891
// MI455X (gfx1250) — 8-bit matrix operand order repaired and checked by reading, compiled, not run on hardware
//
#include <hip/hip_runtime.h>
#include <math.h>
#include <stdint.h>

typedef int v8i __attribute__((ext_vector_type(8)));

#define DIMK    2048
#define HIDDEN  8192
#define TOKENS  8192

static constexpr size_t OFF_W1Q = 0;
static constexpr size_t OFF_W2Q = OFF_W1Q + (size_t)HIDDEN * DIMK;
static constexpr size_t OFF_XQ  = OFF_W2Q + (size_t)DIMK * HIDDEN;
static constexpr size_t OFF_HQ  = OFF_XQ  + (size_t)TOKENS * DIMK;
static constexpr size_t OFF_H   = OFF_HQ  + (size_t)TOKENS * HIDDEN;
static constexpr size_t OFF_RI1 = OFF_H   + (size_t)TOKENS * HIDDEN * 4;
static constexpr size_t OFF_RI2 = OFF_RI1 + (size_t)TOKENS * 4;
static constexpr size_t OFF_P1  = OFF_RI2 + (size_t)TOKENS * 4;
static constexpr size_t OFF_P2  = OFF_P1  + 1024 * 4;
static constexpr size_t OFF_SC  = OFF_P2  + 1024 * 4;

__device__ __forceinline__ int clampi(int v, int lo, int hi) {
    return v < lo ? lo : (v > hi ? hi : v);
}

__global__ void __launch_bounds__(256)
abs_sum_partial(const float* __restrict__ w, size_t n, float* __restrict__ partial) {
    __shared__ float red[256];
    const int t = threadIdx.x;
    float s = 0.f;
    for (size_t i = (size_t)blockIdx.x * 256 + t; i < n; i += (size_t)gridDim.x * 256)
        s += fabsf(w[i]);
    red[t] = s;
    __syncthreads();
    for (int stride = 128; stride > 0; stride >>= 1) {
        if (t < stride) red[t] += red[t + stride];
        __syncthreads();
    }
    if (t == 0) partial[blockIdx.x] = red[0];
}

__global__ void __launch_bounds__(256)
finalize_scale(const float* __restrict__ partial, int np, float invCount,
               float* __restrict__ scaleOut, float* __restrict__ invOut) {
    __shared__ float red[256];
    const int t = threadIdx.x;
    float s = 0.f;
    for (int i = t; i < np; i += 256) s += partial[i];
    red[t] = s;
    __syncthreads();
    for (int stride = 128; stride > 0; stride >>= 1) {
        if (t < stride) red[t] += red[t + stride];
        __syncthreads();
    }
    if (t == 0) {
        float mean = red[0] * invCount;
        float m = fmaxf(mean, 1e-5f);
        scaleOut[0] = 1.0f / m;
        invOut[0]   = m;
    }
}

__global__ void __launch_bounds__(256)
weight_quant_pack(const float* __restrict__ w, const float* __restrict__ scalePtr,
                  uint32_t* __restrict__ Bpack, int N, int K) {
    const float s = scalePtr[0];
    const int ktiles = K >> 6;
    size_t total = ((size_t)N * K) >> 2;
    for (size_t d = (size_t)blockIdx.x * 256 + threadIdx.x; d < total;
         d += (size_t)gridDim.x * 256) {
        size_t tile = d >> 8;
        int r    = (int)(d & 255);
        int lane = r >> 3;
        int vv   = r & 7;
        int nt = (int)(tile / ktiles);
        int kt = (int)(tile % ktiles);
        int n  = nt * 16 + (lane & 15);
        int k0 = kt * 64 + (vv >> 2) * 32 + ((lane >> 4) & 1) * 16 + (vv & 3) * 4;
        const float4 wv = *(const float4*)(w + (size_t)n * K + k0);
        int q0 = clampi((int)rintf(wv.x * s), -1, 1);
        int q1 = clampi((int)rintf(wv.y * s), -1, 1);
        int q2 = clampi((int)rintf(wv.z * s), -1, 1);
        int q3 = clampi((int)rintf(wv.w * s), -1, 1);
        Bpack[d] = (uint32_t)(q0 & 255) | ((uint32_t)(q1 & 255) << 8) |
                   ((uint32_t)(q2 & 255) << 16) | ((uint32_t)(q3 & 255) << 24);
    }
}

__device__ __forceinline__ size_t apack_dword(int row, int kGlobal, int ktiles) {
    int kt = kGlobal >> 6;
    int kk = kGlobal & 63;
    int lane = (row & 15) + (((kk >> 4) & 1) << 4);
    int vv   = (((kk >> 5) & 1) << 2) | ((kk >> 2) & 3);
    size_t tile = (size_t)(row >> 4) * ktiles + kt;
    return tile * 256 + (size_t)lane * 8 + vv;
}

__device__ __forceinline__ uint32_t quant_pack4(float a, float b, float c, float d, float s) {
    int q0 = clampi((int)rintf(a * s), -128, 127);
    int q1 = clampi((int)rintf(b * s), -128, 127);
    int q2 = clampi((int)rintf(c * s), -128, 127);
    int q3 = clampi((int)rintf(d * s), -128, 127);
    return (uint32_t)(q0 & 255) | ((uint32_t)(q1 & 255) << 8) |
           ((uint32_t)(q2 & 255) << 16) | ((uint32_t)(q3 & 255) << 24);
}

__global__ void __launch_bounds__(256)
rmsnorm_quant_pack(const float* __restrict__ x, const float* __restrict__ gamma,
                   uint32_t* __restrict__ Apack, float* __restrict__ rowInv) {
    __shared__ float red[256];
    const int row = blockIdx.x;
    const int t = threadIdx.x;
    const float* xr = x + (size_t)row * DIMK;
    const int k0 = t * 8;

    float4 x0 = *(const float4*)(xr + k0);
    float4 x1 = *(const float4*)(xr + k0 + 4);
    float ss = x0.x * x0.x + x0.y * x0.y + x0.z * x0.z + x0.w * x0.w +
               x1.x * x1.x + x1.y * x1.y + x1.z * x1.z + x1.w * x1.w;
    red[t] = ss;
    __syncthreads();
    for (int stride = 128; stride > 0; stride >>= 1) {
        if (t < stride) red[t] += red[t + stride];
        __syncthreads();
    }
    const float rstd = rsqrtf(red[0] * (1.0f / DIMK) + 1e-6f);
    __syncthreads();

    float4 g0 = *(const float4*)(gamma + k0);
    float4 g1 = *(const float4*)(gamma + k0 + 4);
    float y[8];
    y[0] = x0.x * rstd * g0.x; y[1] = x0.y * rstd * g0.y;
    y[2] = x0.z * rstd * g0.z; y[3] = x0.w * rstd * g0.w;
    y[4] = x1.x * rstd * g1.x; y[5] = x1.y * rstd * g1.y;
    y[6] = x1.z * rstd * g1.z; y[7] = x1.w * rstd * g1.w;
    float am = 0.f;
#pragma unroll
    for (int e = 0; e < 8; ++e) am = fmaxf(am, fabsf(y[e]));
    red[t] = am;
    __syncthreads();
    for (int stride = 128; stride > 0; stride >>= 1) {
        if (t < stride) red[t] = fmaxf(red[t], red[t + stride]);
        __syncthreads();
    }
    const float amax = fmaxf(red[0], 1e-5f);
    const float s = 127.0f / amax;
    if (t == 0) rowInv[row] = amax * (1.0f / 127.0f);

    const int ktiles = DIMK >> 6;
    Apack[apack_dword(row, k0, ktiles)]     = quant_pack4(y[0], y[1], y[2], y[3], s);
    Apack[apack_dword(row, k0 + 4, ktiles)] = quant_pack4(y[4], y[5], y[6], y[7], s);
}

__global__ void __launch_bounds__(256)
absmax_quant_pack_h(const float* __restrict__ h, uint32_t* __restrict__ Apack,
                    float* __restrict__ rowInv) {
    __shared__ float red[256];
    const int row = blockIdx.x;
    const int t = threadIdx.x;
    const float* hr = h + (size_t)row * HIDDEN;
    const int k0 = t * 32;

    float4 v[8];
    float am = 0.f;
#pragma unroll
    for (int e = 0; e < 8; ++e) {
        v[e] = *(const float4*)(hr + k0 + e * 4);
        am = fmaxf(am, fmaxf(fmaxf(fabsf(v[e].x), fabsf(v[e].y)),
                             fmaxf(fabsf(v[e].z), fabsf(v[e].w))));
    }
    red[t] = am;
    __syncthreads();
    for (int stride = 128; stride > 0; stride >>= 1) {
        if (t < stride) red[t] = fmaxf(red[t], red[t + stride]);
        __syncthreads();
    }
    const float amax = fmaxf(red[0], 1e-5f);
    const float s = 127.0f / amax;
    if (t == 0) rowInv[row] = amax * (1.0f / 127.0f);

    const int ktiles = HIDDEN >> 6;
#pragma unroll
    for (int e = 0; e < 8; ++e)
        Apack[apack_dword(row, k0 + e * 4, ktiles)] =
            quant_pack4(v[e].x, v[e].y, v[e].z, v[e].w, s);
}

template <bool GELU>
__global__ void __launch_bounds__(128)
bit_gemm_iu8(const uint8_t* __restrict__ Apack, const uint8_t* __restrict__ Bpack,
             const float* __restrict__ rowInv, const float* __restrict__ wInvPtr,
             float* __restrict__ Cout, int M, int N, int K) {
    const int lane = threadIdx.x & 31;
    const int wave = threadIdx.x >> 5;
    const int tilesN = N >> 6;
    const int tileId = blockIdx.x * 4 + wave;
    const int wm = tileId / tilesN;
    const int wn = tileId % tilesN;
    const int ktiles = K >> 6;

    const uint8_t* aBase = Apack + ((size_t)(wm * 4) * ktiles) * 1024 + (size_t)lane * 32;
    const uint8_t* bBase = Bpack + ((size_t)(wn * 4) * ktiles) * 1024 + (size_t)lane * 32;

    v8i c[4][4] = {};
    for (int kt = 0; kt < ktiles; ++kt) {
        v8i a[4], b[4];
#pragma unroll
        for (int i = 0; i < 4; ++i)
            a[i] = *(const v8i*)(aBase + ((size_t)i * ktiles + kt) * 1024);
#pragma unroll
        for (int j = 0; j < 4; ++j)
            b[j] = *(const v8i*)(bBase + ((size_t)j * ktiles + kt) * 1024);
#pragma unroll
        for (int i = 0; i < 4; ++i)
#pragma unroll
            for (int j = 0; j < 4; ++j)
                c[i][j] = __builtin_amdgcn_wmma_i32_16x16x64_iu8(
                    true, a[i], true, b[j], c[i][j], false, false);
    }

    const float wInv = wInvPtr[0];
    const int mLane = (lane >> 4) * 8;
    const int nLane = lane & 15;
#pragma unroll
    for (int i = 0; i < 4; ++i) {
        const int mBase = wm * 64 + i * 16 + mLane;
#pragma unroll
        for (int j = 0; j < 4; ++j) {
            const int nIdx = wn * 64 + j * 16 + nLane;
#pragma unroll
            for (int r = 0; r < 8; ++r) {
                const int m = mBase + r;
                float val = (float)c[i][j][r] * rowInv[m] * wInv;
                if (GELU)
                    val = 0.5f * val * (1.0f + erff(val * 0.70710678118654752440f));
                Cout[(size_t)m * N + nIdx] = val;
            }
        }
    }
}

extern "C" void kernel_launch(void* const* d_in, const int* in_sizes, int n_in,
                              void* d_out, int out_size, void* d_ws, size_t ws_size,
                              hipStream_t stream) {
    const float* x     = (const float*)d_in[0];
    const float* w1    = (const float*)d_in[1];
    const float* w2    = (const float*)d_in[2];
    const float* gamma = (const float*)d_in[3];
    float* out = (float*)d_out;

    uint8_t* ws = (uint8_t*)d_ws;
    uint8_t*  w1q = ws + OFF_W1Q;
    uint8_t*  w2q = ws + OFF_W2Q;
    uint8_t*  xq  = ws + OFF_XQ;
    uint8_t*  hq  = ws + OFF_HQ;
    float*    h   = (float*)(ws + OFF_H);
    float*    ri1 = (float*)(ws + OFF_RI1);
    float*    ri2 = (float*)(ws + OFF_RI2);
    float*    p1  = (float*)(ws + OFF_P1);
    float*    p2  = (float*)(ws + OFF_P2);
    float*    sc  = (float*)(ws + OFF_SC);

    const size_t nW1 = (size_t)HIDDEN * DIMK;
    const size_t nW2 = (size_t)DIMK * HIDDEN;

    abs_sum_partial<<<1024, 256, 0, stream>>>(w1, nW1, p1);
    finalize_scale<<<1, 256, 0, stream>>>(p1, 1024, 1.0f / (float)nW1, sc + 0, sc + 1);
    abs_sum_partial<<<1024, 256, 0, stream>>>(w2, nW2, p2);
    finalize_scale<<<1, 256, 0, stream>>>(p2, 1024, 1.0f / (float)nW2, sc + 2, sc + 3);

    weight_quant_pack<<<(int)(nW1 / 4 / 256), 256, 0, stream>>>(
        w1, sc + 0, (uint32_t*)w1q, HIDDEN, DIMK);
    weight_quant_pack<<<(int)(nW2 / 4 / 256), 256, 0, stream>>>(
        w2, sc + 2, (uint32_t*)w2q, DIMK, HIDDEN);

    rmsnorm_quant_pack<<<TOKENS, 256, 0, stream>>>(x, gamma, (uint32_t*)xq, ri1);

    {
        int waves = (TOKENS / 64) * (HIDDEN / 64);
        bit_gemm_iu8<true><<<waves / 4, 128, 0, stream>>>(
            xq, w1q, ri1, sc + 1, h, TOKENS, HIDDEN, DIMK);
    }

    absmax_quant_pack_h<<<TOKENS, 256, 0, stream>>>(h, (uint32_t*)hq, ri2);

    {
        int waves = (TOKENS / 64) * (DIMK / 64);
        bit_gemm_iu8<false><<<waves / 4, 128, 0, stream>>>(
            hq, w2q, ri2, sc + 3, out, TOKENS, DIMK, HIDDEN);
    }
}
